// SpatioTemporalGNN_30580167147641
// MI455X (gfx1250) — matrix operand order repaired and checked by reading, not compiled, not run on hardware
//
#include <hip/hip_runtime.h>
#include <hip/hip_bf16.h>

typedef __bf16 bf16;
typedef __attribute__((ext_vector_type(16))) __bf16 v16bf;
typedef __attribute__((ext_vector_type(8)))  __bf16 v8bf;
typedef __attribute__((ext_vector_type(8)))  float  v8f;

// ---------------------------------------------------------------------------
// Utility fills
// ---------------------------------------------------------------------------
__global__ void fill_i32(int* __restrict__ p, int v, int n) {
    int i = blockIdx.x * blockDim.x + threadIdx.x;
    if (i < n) p[i] = v;
}

__global__ void fill_f32(float* __restrict__ p, float v, int n) {
    int i = blockIdx.x * blockDim.x + threadIdx.x;
    if (i < n) p[i] = v;
}

// ---------------------------------------------------------------------------
// Degree count (in-degree over dst) and dinv = rsqrt(deg + 1 self loop)
// ---------------------------------------------------------------------------
__global__ void edge_deg(const int* __restrict__ dst, int* __restrict__ deg, int E) {
    int e = blockIdx.x * blockDim.x + threadIdx.x;
    if (e < E) atomicAdd(&deg[dst[e]], 1);
}

__global__ void deg_inv(const int* __restrict__ deg, float* __restrict__ dinv, int n) {
    int i = blockIdx.x * blockDim.x + threadIdx.x;
    if (i < n) dinv[i] = rsqrtf((float)(deg[i] + 1));  // +1 = self loop, >= 1
}

// ---------------------------------------------------------------------------
// WMMA GEMM:  C(Nrows x HOUT) = A(Nrows x 128) * B(128 x HOUT)
// WT=false: W row-major (K=128 rows, HOUT cols)   -> B[k][c] = W[k*HOUT+c]
// WT=true : W is (HOUT rows, 128 cols), use W^T   -> B[k][c] = W[c*128+k]
// Block: 256 threads = 8 waves. Block tile 32 rows x 128 cols.
// Wave tile 16 rows x 32 cols: one A fragment shared by two WMMAs.
// K=128 -> 8 x v_wmma_f32_16x16x32_bf16 per wave.
// HOUT is compile-time so all store offsets fold into immediates.
// ---------------------------------------------------------------------------
template <bool WT, int HOUT>
__global__ __launch_bounds__(256) void gemm128_wmma(const float* __restrict__ A,
                                                    const float* __restrict__ W,
                                                    float* __restrict__ C,
                                                    int Nrows) {
    constexpr int K = 128;
    __shared__ __align__(16) bf16 As[32][K + 8];   // +8 halfs pad vs bank conflicts
    __shared__ __align__(16) bf16 Ws[128][K + 8];  // column-major slice: Ws[c][k]

    const int tid = threadIdx.x;
    const int m0  = blockIdx.x * 32;
    const int c0  = blockIdx.y * 128;

    // Stage A tile (32 x 128) as bf16
    for (int i = tid; i < 32 * K; i += 256) {
        int r = i >> 7, k = i & (K - 1);
        int gr = m0 + r;
        float v = (gr < Nrows) ? A[(size_t)gr * K + k] : 0.0f;
        As[r][k] = (bf16)v;
    }
    // Stage W slice (128 cols x 128 k) as bf16, column-major in LDS
    for (int i = tid; i < 128 * K; i += 256) {
        int c = i >> 7, k = i & (K - 1);
        float v = WT ? W[(size_t)(c0 + c) * K + k]
                     : W[(size_t)k * HOUT + (c0 + c)];
        Ws[c][k] = (bf16)v;
    }
    __syncthreads();

    const int wv   = tid >> 5;
    const int lane = tid & 31;
    const int mi   = wv >> 2;   // 0..1 row tile (16 rows each)
    const int ci   = wv & 3;    // 0..3 col tile (32 cols each)
    const int rsel = lane >> 4; // half-wave select
    const int l15  = lane & 15;

    const bf16* arow  = As[mi * 16 + l15];       // A: one matrix row per lane
    const bf16* bcol0 = Ws[ci * 32 + l15];       // B: one matrix col per lane
    const bf16* bcol1 = Ws[ci * 32 + 16 + l15];

    v8f acc0 = {0.f, 0.f, 0.f, 0.f, 0.f, 0.f, 0.f, 0.f};
    v8f acc1 = {0.f, 0.f, 0.f, 0.f, 0.f, 0.f, 0.f, 0.f};

#pragma unroll
    for (int kk = 0; kk < K; kk += 32) {
        // A 16x32 fragment: lanes 0-15 hold K {0..7,16..23}, lanes 16-31 {8..15,24..31}
        const int aoff = kk + rsel * 8;
        v8bf alo = *(const v8bf*)(arow + aoff);
        v8bf ahi = *(const v8bf*)(arow + aoff + 16);
        v16bf a = __builtin_shufflevector(alo, ahi, 0, 1, 2, 3, 4, 5, 6, 7,
                                          8, 9, 10, 11, 12, 13, 14, 15);
        // B 32x16 fragments: lanes 0-15 hold K {0..7,16..23}, lanes 16-31 {8..15,24..31}
        const int boff = kk + rsel * 8;
        v8bf b0l = *(const v8bf*)(bcol0 + boff);
        v8bf b0h = *(const v8bf*)(bcol0 + boff + 16);
        v16bf b0 = __builtin_shufflevector(b0l, b0h, 0, 1, 2, 3, 4, 5, 6, 7,
                                           8, 9, 10, 11, 12, 13, 14, 15);
        v8bf b1l = *(const v8bf*)(bcol1 + boff);
        v8bf b1h = *(const v8bf*)(bcol1 + boff + 16);
        v16bf b1 = __builtin_shufflevector(b1l, b1h, 0, 1, 2, 3, 4, 5, 6, 7,
                                           8, 9, 10, 11, 12, 13, 14, 15);
        acc0 = __builtin_amdgcn_wmma_f32_16x16x32_bf16(
            false, a, false, b0, (short)0, acc0, false, false);
        acc1 = __builtin_amdgcn_wmma_f32_16x16x32_bf16(
            false, a, false, b1, (short)0, acc1, false, false);
    }

    // D layout: VGPR r -> row (r + 8*rsel), col = l15 within each 16-col tile
    const int gcol = c0 + ci * 32 + l15;
    const int grow = m0 + mi * 16 + rsel * 8;
    float* cp = C + (size_t)grow * HOUT + gcol;
    if (m0 + 32 <= Nrows) {  // uniform fast path: offsets are immediates
#pragma unroll
        for (int r = 0; r < 8; ++r) {
            cp[r * HOUT]      = acc0[r];
            cp[r * HOUT + 16] = acc1[r];
        }
    } else {
#pragma unroll
        for (int r = 0; r < 8; ++r) {
            if (grow + r < Nrows) {
                cp[r * HOUT]      = acc0[r];
                cp[r * HOUT + 16] = acc1[r];
            }
        }
    }
}

// ---------------------------------------------------------------------------
// Edge scatter: agg[dst] += xw[src] * dinv[src]*dinv[dst]
// One wave per edge, float4 per lane, single-instruction f32 atomics.
// ---------------------------------------------------------------------------
__global__ __launch_bounds__(256) void edge_scatter(const int* __restrict__ src,
                                                    const int* __restrict__ dst,
                                                    const float* __restrict__ dinv,
                                                    const float* __restrict__ xw,
                                                    float* __restrict__ agg, int E) {
    const int lane = threadIdx.x & 31;
    const int e = blockIdx.x * 8 + (threadIdx.x >> 5);
    if (e >= E) return;
    const int s = src[e], d = dst[e];
    const float nrm = dinv[s] * dinv[d];
    const float4 v = *(const float4*)(xw + (size_t)s * 128 + lane * 4);
    float* p = agg + (size_t)d * 128 + lane * 4;
    unsafeAtomicAdd(p + 0, v.x * nrm);
    unsafeAtomicAdd(p + 1, v.y * nrm);
    unsafeAtomicAdd(p + 2, v.z * nrm);
    unsafeAtomicAdd(p + 3, v.w * nrm);
}

// ---------------------------------------------------------------------------
// GCN finalize: y = [resid +] relu(agg + dinv[n]^2 * xw + b)
// (self-loop term folded in here; resid==nullptr for layer 1)
// ---------------------------------------------------------------------------
__global__ void gcn_finalize(const float* __restrict__ agg, const float* __restrict__ xw,
                             const float* __restrict__ dinv, const float* __restrict__ b,
                             const float* __restrict__ resid, float* __restrict__ y, int NH) {
    int i = blockIdx.x * blockDim.x + threadIdx.x;
    if (i >= NH) return;
    const int n = i >> 7, f = i & 127;
    const float di = dinv[n];
    float v = fmaxf(agg[i] + di * di * xw[i] + b[f], 0.0f);
    if (resid) v += resid[i];
    y[i] = v;
}

// ---------------------------------------------------------------------------
// LayerNorm over H=128: one wave per node, shfl_xor reductions (wave32)
// ---------------------------------------------------------------------------
__global__ __launch_bounds__(256) void layernorm_k(const float* __restrict__ x,
                                                   const float* __restrict__ gamma,
                                                   const float* __restrict__ beta,
                                                   float* __restrict__ y, int Nn) {
    const int lane = threadIdx.x & 31;
    const int n = blockIdx.x * 8 + (threadIdx.x >> 5);
    if (n >= Nn) return;
    const float4 v = *(const float4*)(x + (size_t)n * 128 + lane * 4);
    float s = v.x + v.y + v.z + v.w;
    for (int o = 16; o; o >>= 1) s += __shfl_xor(s, o, 32);
    const float mu = s * (1.0f / 128.0f);
    float4 d = {v.x - mu, v.y - mu, v.z - mu, v.w - mu};
    float q = d.x * d.x + d.y * d.y + d.z * d.z + d.w * d.w;
    for (int o = 16; o; o >>= 1) q += __shfl_xor(q, o, 32);
    const float inv = rsqrtf(q * (1.0f / 128.0f) + 1e-5f);
    const float4 g = *(const float4*)(gamma + lane * 4);
    const float4 b = *(const float4*)(beta + lane * 4);
    float4 r = {d.x * inv * g.x + b.x, d.y * inv * g.y + b.y,
                d.z * inv * g.z + b.z, d.w * inv * g.w + b.w};
    *(float4*)(y + (size_t)n * 128 + lane * 4) = r;
}

// ---------------------------------------------------------------------------
// GRU gates (r,z,n order), biases folded in here; writes both output copies.
// ---------------------------------------------------------------------------
__global__ void gru_gate(const float* __restrict__ gi, const float* __restrict__ gh,
                         const float* __restrict__ bih, const float* __restrict__ bhh,
                         const float* __restrict__ hprev, float* __restrict__ out, int NH) {
    int i = blockIdx.x * blockDim.x + threadIdx.x;
    if (i >= NH) return;
    const int n = i >> 7, f = i & 127;
    const size_t base = (size_t)n * 384;
    const float ir  = gi[base + f]       + bih[f];
    const float iz  = gi[base + 128 + f] + bih[128 + f];
    const float in_ = gi[base + 256 + f] + bih[256 + f];
    const float hr  = gh[base + f]       + bhh[f];
    const float hz  = gh[base + 128 + f] + bhh[128 + f];
    const float hn  = gh[base + 256 + f] + bhh[256 + f];
    const float r = 1.0f / (1.0f + __expf(-(ir + hr)));
    const float z = 1.0f / (1.0f + __expf(-(iz + hz)));
    const float ng = tanhf(in_ + r * hn);
    const float nh = (1.0f - z) * ng + z * hprev[i];
    out[i] = nh;
    out[(size_t)NH + i] = nh;  // second tuple output (1,N,H)
}

// ---------------------------------------------------------------------------
// Launch
// ---------------------------------------------------------------------------
extern "C" void kernel_launch(void* const* d_in, const int* in_sizes, int n_in,
                              void* d_out, int out_size, void* d_ws, size_t ws_size,
                              hipStream_t stream) {
    const float* x      = (const float*)d_in[0];
    const int*   ei     = (const int*)d_in[1];
    const float* hidden = (const float*)d_in[2];
    const float* W1     = (const float*)d_in[3];
    const float* b1     = (const float*)d_in[4];
    const float* W2     = (const float*)d_in[5];
    const float* b2     = (const float*)d_in[6];
    const float* gamma  = (const float*)d_in[7];
    const float* beta   = (const float*)d_in[8];
    const float* W_ih   = (const float*)d_in[9];
    const float* W_hh   = (const float*)d_in[10];
    const float* b_ih   = (const float*)d_in[11];
    const float* b_hh   = (const float*)d_in[12];
    float* out = (float*)d_out;

    const int Nn = in_sizes[0] / 128;
    const int E  = in_sizes[1] / 2;
    const int NH = Nn * 128;
    const int* src = ei;
    const int* dst = ei + E;

    // Workspace layout (regions reused across phases):
    //   degi (N int) | dinv (N f32) | regA = 3*NH f32 {xw, agg | later gi}
    //                                | regB = 3*NH f32 {h1     | later gh}
    char* wsc = (char*)d_ws;
    const size_t nAl = (((size_t)Nn * 4) + 255) & ~(size_t)255;
    int*   degi = (int*)wsc;
    float* dinv = (float*)(wsc + nAl);
    float* regA = (float*)(wsc + 2 * nAl);
    float* regB = regA + (size_t)3 * NH;
    float* xw  = regA;            // GEMM outputs (layer1 then layer2, reused)
    float* agg = regA + NH;       // scatter accumulator (reused)
    float* gi  = regA;            // reuses xw+agg after GCN layers are done
    float* h1  = regB;            // GCN hidden (dead after layernorm)
    float* gh  = regB;            // reuses h1's region
    float* hn  = out + NH;        // LayerNorm output staged in d_out's 2nd half,
                                  // overwritten by gru_gate at the end

    const int T = 256;
    const dim3 gN((Nn + T - 1) / T);
    const dim3 gE((E + T - 1) / T);
    const dim3 gNH((NH + T - 1) / T);
    const dim3 gEdgeW((E + 7) / 8);              // one wave32 per edge
    const dim3 gLN((Nn + 7) / 8);                // one wave32 per node
    const dim3 gGemmH((Nn + 31) / 32, 1);        // HOUT=128: whole W per block
    const dim3 gGemm3H((Nn + 31) / 32, 3);       // HOUT=384: 3 column slabs

    // Degrees and symmetric normalization
    fill_i32<<<gN, T, 0, stream>>>(degi, 0, Nn);
    edge_deg<<<gE, T, 0, stream>>>(dst, degi, E);
    deg_inv<<<gN, T, 0, stream>>>(degi, dinv, Nn);

    // GCN layer 1: xw = x@W1 ; aggregate ; h1 = relu(agg + dinv^2*xw + b1)
    gemm128_wmma<false, 128><<<gGemmH, T, 0, stream>>>(x, W1, xw, Nn);
    fill_f32<<<gNH, T, 0, stream>>>(agg, 0.0f, NH);
    edge_scatter<<<gEdgeW, T, 0, stream>>>(src, dst, dinv, xw, agg, E);
    gcn_finalize<<<gNH, T, 0, stream>>>(agg, xw, dinv, b1, nullptr, h1, NH);

    // GCN layer 2 (residual): h1 += relu(agg2 + dinv^2*(h1@W2) + b2)
    gemm128_wmma<false, 128><<<gGemmH, T, 0, stream>>>(h1, W2, xw, Nn);
    fill_f32<<<gNH, T, 0, stream>>>(agg, 0.0f, NH);
    edge_scatter<<<gEdgeW, T, 0, stream>>>(src, dst, dinv, xw, agg, E);
    gcn_finalize<<<gNH, T, 0, stream>>>(agg, xw, dinv, b2, h1, h1, NH);

    // LayerNorm
    layernorm_k<<<gLN, T, 0, stream>>>(h1, gamma, beta, hn, Nn);

    // GRU: gi = hn@W_ih^T ; gh = h_prev@W_hh^T ; gates
    gemm128_wmma<true, 384><<<gGemm3H, T, 0, stream>>>(hn, W_ih, gi, Nn);
    gemm128_wmma<true, 384><<<gGemm3H, T, 0, stream>>>(hidden, W_hh, gh, Nn);
    gru_gate<<<gNH, T, 0, stream>>>(gi, gh, b_ih, b_hh, hidden, out, NH);
}
